// Seq2Seq_18760417149688
// MI455X (gfx1250) — matrix operand order repaired and checked by reading, not compiled, not run on hardware
//
#include <hip/hip_runtime.h>

typedef __bf16 bf16_t;
typedef float  v8f   __attribute__((ext_vector_type(8)));
typedef bf16_t v16bf __attribute__((ext_vector_type(16)));
typedef bf16_t v8bf  __attribute__((ext_vector_type(8)));

__device__ __forceinline__ bf16_t f2bf(float f) {
  unsigned u = __builtin_bit_cast(unsigned, f);
  unsigned r = u + 0x7FFFu + ((u >> 16) & 1u);
  unsigned short h = (unsigned short)(r >> 16);
  return __builtin_bit_cast(bf16_t, h);
}
__device__ __forceinline__ float sigm(float x) { return 1.0f / (1.0f + __expf(-x)); }
__device__ __forceinline__ v16bf cat8(v8bf a, v8bf b) {
  return __builtin_shufflevector(a, b, 0,1,2,3,4,5,6,7,8,9,10,11,12,13,14,15);
}
__device__ __forceinline__ v8f vzero8() { v8f z = {0.f,0.f,0.f,0.f,0.f,0.f,0.f,0.f}; return z; }

// ---------------------------------------------------------------------------
// Conversion kernels (f32 -> bf16, optional K pad / transpose / dest offset)
// ---------------------------------------------------------------------------
__global__ void cvt_rows_k(const float* __restrict__ in, bf16_t* __restrict__ out,
                           int Nrows, int Ka, int Kp, int dstOff, int ld) {
  long i = (long)blockIdx.x * 256 + threadIdx.x;
  long tot = (long)Nrows * Kp;
  if (i >= tot) return;
  int n = (int)(i / Kp), k = (int)(i % Kp);
  float v = (k < Ka) ? in[(long)n * Ka + k] : 0.f;
  out[(long)n * ld + dstOff + k] = f2bf(v);
}

// in: [K][N] f32  ->  out: [Np][K] bf16 (rows n>=N zero padded)
__global__ void cvt_T_k(const float* __restrict__ in, bf16_t* __restrict__ out,
                        int K, int N, int Np) {
  long i = (long)blockIdx.x * 256 + threadIdx.x;
  long tot = (long)Np * K;
  if (i >= tot) return;
  int n = (int)(i / K), k = (int)(i % K);
  float v = (n < N) ? in[(long)k * N + n] : 0.f;
  out[i] = f2bf(v);
}

// x (B,T,40) f32 -> (B,T,64) bf16 zero padded
__global__ void cvt_x_k(const float* __restrict__ in, bf16_t* __restrict__ out, long BT) {
  long i = (long)blockIdx.x * 256 + threadIdx.x;
  if (i >= BT * 64) return;
  long bt = i >> 6; int k = (int)(i & 63);
  float v = (k < 40) ? in[bt * 40 + k] : 0.f;
  out[i] = f2bf(v);
}

// ---------------------------------------------------------------------------
// Generic bf16 WMMA GEMM: out[m][n] = sum_k A[row(m)][k] * W[n][k] + bias[n]
// A row element offset: (m % Bdim)*sB + (m / Bdim)*sT
// One 16x16 tile per wave, 8 waves / block, K multiple of 32, M,N mult of 16.
// ---------------------------------------------------------------------------
__global__ __launch_bounds__(256) void gemm_xw_k(
    const bf16_t* __restrict__ A, const bf16_t* __restrict__ W,
    const float* __restrict__ bias, float* __restrict__ out,
    int tilesN, int totalTiles, int N, int Kp, int Bdim, long sB, long sT) {
  int wave = threadIdx.x >> 5;
  int gid = blockIdx.x * 8 + wave;
  if (gid >= totalTiles) return;
  int lane = threadIdx.x & 31;
  int half = lane >> 4, l15 = lane & 15;
  int tm = gid / tilesN, tn = gid - tm * tilesN;
  int mrow = tm * 16 + l15;
  const bf16_t* ap = A + (long)(mrow % Bdim) * sB + (long)(mrow / Bdim) * sT;
  const bf16_t* wp = W + (long)(tn * 16 + l15) * Kp + half * 8;
  v8f acc = vzero8();
  for (int k0 = 0; k0 < Kp; k0 += 32) {
    v8bf a0 = *(const v8bf*)(ap + k0 + half * 8);
    v8bf a1 = *(const v8bf*)(ap + k0 + 16 + half * 8);
    v16bf av = cat8(a0, a1);
    v16bf bv = cat8(*(const v8bf*)(wp + k0), *(const v8bf*)(wp + k0 + 16));
    acc = __builtin_amdgcn_wmma_f32_16x16x32_bf16(false, av, false, bv, (short)0, acc, false, false);
  }
  int n = tn * 16 + l15;
  float bb = bias[n];
  long base = (long)(tm * 16 + half * 8) * N + n;
#pragma unroll
  for (int r = 0; r < 8; ++r) out[base + (long)r * N] = acc[r] + bb;
}

// ---------------------------------------------------------------------------
// Persistent recurrent LSTM (one workgroup per direction).
// gates(32x1024) = h(32x256 bf16, LDS) @ Whh^T via WMMA; xW (incl bias) added
// in the fp32 elementwise phase. Variable lengths; dir1 = reversed-padded.
// hout: (32, T, 512) bf16, dir d in columns [d*256, d*256+256).
// ---------------------------------------------------------------------------
__global__ __launch_bounds__(1024) void lstm_rec_k(
    const float* __restrict__ xW,    // (2, T, 32, 1024)
    const bf16_t* __restrict__ Whh,  // (2, 1024, 256)
    const int* __restrict__ lens, int shift,
    bf16_t* __restrict__ hout, int T) {
  __shared__ float  gates[32 * 1024];   // 128 KB
  __shared__ float  cbuf[32 * 256];     // 32 KB
  __shared__ bf16_t hbuf[32 * 256];     // 16 KB
  int dir = blockIdx.x;
  const float*  xw  = xW  + (long)dir * T * 32 * 1024;
  const bf16_t* whh = Whh + (long)dir * 1024 * 256;
  int tid = threadIdx.x, lane = tid & 31, wave = tid >> 5;
  int half = lane >> 4, l15 = lane & 15;
  for (int i = tid; i < 32 * 256; i += 1024) { cbuf[i] = 0.f; hbuf[i] = f2bf(0.f); }
  __syncthreads();
  int mtile = wave >> 4;          // 0..1
  int nt0 = (wave & 15) * 4;      // 4 n-tiles per wave
  int mrow = mtile * 16 + l15;    // batch row 0..31
  for (int s = 0; s < T; ++s) {
    v8f acc[4];
#pragma unroll
    for (int t = 0; t < 4; ++t) acc[t] = vzero8();
    for (int k0 = 0; k0 < 256; k0 += 32) {
      v8bf a0 = *(const v8bf*)&hbuf[mrow * 256 + k0 + half * 8];
      v8bf a1 = *(const v8bf*)&hbuf[mrow * 256 + k0 + 16 + half * 8];
      v16bf av = cat8(a0, a1);
#pragma unroll
      for (int t = 0; t < 4; ++t) {
        int n = (nt0 + t) * 16 + l15;
        v16bf bv = cat8(*(const v8bf*)(whh + (long)n * 256 + half * 8 + k0), *(const v8bf*)(whh + (long)n * 256 + 16 + half * 8 + k0));
        acc[t] = __builtin_amdgcn_wmma_f32_16x16x32_bf16(false, av, false, bv, (short)0, acc[t], false, false);
      }
    }
#pragma unroll
    for (int t = 0; t < 4; ++t) {
      int n = (nt0 + t) * 16 + l15;
#pragma unroll
      for (int r = 0; r < 8; ++r)
        gates[(mtile * 16 + half * 8 + r) * 1024 + n] = acc[t][r];
    }
    __syncthreads();
#pragma unroll
    for (int r = 0; r < 8; ++r) {
      int idx = tid + r * 1024;           // 0..8191
      int b = idx >> 8, j = idx & 255;
      int len = lens[b] >> shift;
      if (s < len) {
        int tio = dir ? (len - 1 - s) : s;
        long xo = ((long)tio * 32 + b) * 1024 + j;
        float gi = gates[b * 1024 + j]        + xw[xo];
        float gf = gates[b * 1024 + 256 + j]  + xw[xo + 256];
        float gg = gates[b * 1024 + 512 + j]  + xw[xo + 512];
        float go = gates[b * 1024 + 768 + j]  + xw[xo + 768];
        float c = sigm(gf) * cbuf[b * 256 + j] + sigm(gi) * tanhf(gg);
        float h = sigm(go) * tanhf(c);
        cbuf[b * 256 + j] = c;
        bf16_t hb = f2bf(h);
        hbuf[b * 256 + j] = hb;
        hout[((long)b * T + tio) * 512 + dir * 256 + j] = hb;
      }
    }
    __syncthreads();
  }
}

// ---------------------------------------------------------------------------
// Persistent attention decoder: one 1024-thread workgroup, 199 steps.
// NOTE: the reference builds the attention mask from the ORIGINAL lens
// (1024..2048) against t in [0,256), so the mask is all-true; we replicate
// that exactly by comparing t against lens[b] directly.
// ---------------------------------------------------------------------------
__global__ __launch_bounds__(1024) void decoder_k(
    const int* __restrict__ text, const bf16_t* __restrict__ embbf,
    const float* __restrict__ keys, const float* __restrict__ values,
    const int* __restrict__ lens,
    const bf16_t* __restrict__ Wc1, const float* __restrict__ d1b,
    const bf16_t* __restrict__ Wc2, const float* __restrict__ d2b,
    const bf16_t* __restrict__ WoutT, const float* __restrict__ outb,
    float* __restrict__ preds, float* __restrict__ attns) {
  __shared__ float  scratch[32 * 1024];  // gates1 | gates2(first 512) ; ebuf at +32*512
  __shared__ bf16_t A1[32 * 640];        // [emb(256) | context(128) | h1(256)]
  __shared__ bf16_t A2[32 * 384];        // [h1(256) | h2(128)]
  __shared__ bf16_t Ap[32 * 256];        // [h2(128) | context(128)]
  __shared__ float  c1[32 * 256];
  __shared__ float  c2[32 * 128];
  __shared__ float  h2s[32 * 128];
  int tid = threadIdx.x, lane = tid & 31, wave = tid >> 5;
  int half = lane >> 4, l15 = lane & 15;
  bf16_t bz = f2bf(0.f);
  for (int i = tid; i < 32 * 640; i += 1024) A1[i] = bz;
  for (int i = tid; i < 32 * 384; i += 1024) A2[i] = bz;
  for (int i = tid; i < 32 * 256; i += 1024) { Ap[i] = bz; c1[i] = 0.f; }
  for (int i = tid; i < 32 * 128; i += 1024) { c2[i] = 0.f; h2s[i] = 0.f; }
  __syncthreads();
  float* ebuf = scratch + 32 * 512;

  for (int s = 0; s < 199; ++s) {
    // embed token into A1[:,0:256]
    for (int i = tid; i < 32 * 256; i += 1024) {
      int b = i >> 8, k = i & 255;
      int tok = text[b * 200 + s];
      A1[b * 640 + k] = embbf[tok * 256 + k];
    }
    __syncthreads();
    // d1 gates: (32x1024) = A1(32x640) @ Wc1^T
    {
      int mtile = wave >> 4, nt0 = (wave & 15) * 4;
      int mrow = mtile * 16 + l15;
      v8f acc[4];
#pragma unroll
      for (int t = 0; t < 4; ++t) acc[t] = vzero8();
      for (int k0 = 0; k0 < 640; k0 += 32) {
        v8bf a0 = *(const v8bf*)&A1[mrow * 640 + k0 + half * 8];
        v8bf a1 = *(const v8bf*)&A1[mrow * 640 + k0 + 16 + half * 8];
        v16bf av = cat8(a0, a1);
#pragma unroll
        for (int t = 0; t < 4; ++t) {
          int n = (nt0 + t) * 16 + l15;
          v16bf bv = cat8(*(const v8bf*)(Wc1 + (long)n * 640 + half * 8 + k0), *(const v8bf*)(Wc1 + (long)n * 640 + 16 + half * 8 + k0));
          acc[t] = __builtin_amdgcn_wmma_f32_16x16x32_bf16(false, av, false, bv, (short)0, acc[t], false, false);
        }
      }
#pragma unroll
      for (int t = 0; t < 4; ++t) {
        int n = (nt0 + t) * 16 + l15;
#pragma unroll
        for (int r = 0; r < 8; ++r)
          scratch[(mtile * 16 + half * 8 + r) * 1024 + n] = acc[t][r];
      }
    }
    __syncthreads();
    // LSTM cell 1
#pragma unroll
    for (int r = 0; r < 8; ++r) {
      int idx = tid + r * 1024;
      int b = idx >> 8, j = idx & 255;
      float gi = scratch[b * 1024 + j]       + d1b[j];
      float gf = scratch[b * 1024 + 256 + j] + d1b[256 + j];
      float gg = scratch[b * 1024 + 512 + j] + d1b[512 + j];
      float go = scratch[b * 1024 + 768 + j] + d1b[768 + j];
      float c = sigm(gf) * c1[b * 256 + j] + sigm(gi) * tanhf(gg);
      float h = sigm(go) * tanhf(c);
      c1[b * 256 + j] = c;
      bf16_t hb = f2bf(h);
      A1[b * 640 + 384 + j] = hb;
      A2[b * 384 + j] = hb;
    }
    __syncthreads();
    // d2 gates: (32x512) = A2(32x384) @ Wc2^T  (64 tiles, 2 per wave)
    {
      int mtile = wave >> 4, nt0 = (wave & 15) * 2;
      int mrow = mtile * 16 + l15;
      v8f acc[2];
      acc[0] = vzero8(); acc[1] = vzero8();
      for (int k0 = 0; k0 < 384; k0 += 32) {
        v8bf a0 = *(const v8bf*)&A2[mrow * 384 + k0 + half * 8];
        v8bf a1 = *(const v8bf*)&A2[mrow * 384 + k0 + 16 + half * 8];
        v16bf av = cat8(a0, a1);
#pragma unroll
        for (int t = 0; t < 2; ++t) {
          int n = (nt0 + t) * 16 + l15;
          v16bf bv = cat8(*(const v8bf*)(Wc2 + (long)n * 384 + half * 8 + k0), *(const v8bf*)(Wc2 + (long)n * 384 + 16 + half * 8 + k0));
          acc[t] = __builtin_amdgcn_wmma_f32_16x16x32_bf16(false, av, false, bv, (short)0, acc[t], false, false);
        }
      }
#pragma unroll
      for (int t = 0; t < 2; ++t) {
        int n = (nt0 + t) * 16 + l15;
#pragma unroll
        for (int r = 0; r < 8; ++r)
          scratch[(mtile * 16 + half * 8 + r) * 512 + n] = acc[t][r];
      }
    }
    __syncthreads();
    // LSTM cell 2
#pragma unroll
    for (int r = 0; r < 4; ++r) {
      int idx = tid + r * 1024;
      int b = idx >> 7, j = idx & 127;
      float gi = scratch[b * 512 + j]       + d2b[j];
      float gf = scratch[b * 512 + 128 + j] + d2b[128 + j];
      float gg = scratch[b * 512 + 256 + j] + d2b[256 + j];
      float go = scratch[b * 512 + 384 + j] + d2b[384 + j];
      float c = sigm(gf) * c2[b * 128 + j] + sigm(gi) * tanhf(gg);
      float h = sigm(go) * tanhf(c);
      c2[b * 128 + j] = c;
      h2s[b * 128 + j] = h;
      bf16_t hb = f2bf(h);
      A2[b * 384 + 256 + j] = hb;
      Ap[b * 256 + j] = hb;
    }
    __syncthreads();
    // energy[b][t] = keys[b][t][:] . h2[b][:]
#pragma unroll
    for (int r = 0; r < 8; ++r) {
      int idx = tid + r * 1024;
      int b = idx >> 8, t = idx & 255;
      const float* kp = keys + ((long)b * 256 + t) * 128;
      float a = 0.f;
      for (int k = 0; k < 128; ++k) a += kp[k] * h2s[b * 128 + k];
      ebuf[b * 256 + t] = a;
    }
    __syncthreads();
    // softmax per batch row (wave w handles b = w); mask uses ORIGINAL lens
    // (always >= 1024 > 255) to match reference semantics exactly.
    {
      int b = wave;
      int lenb = lens[b];
      float ev[8];
      float m = -1e30f;
#pragma unroll
      for (int r = 0; r < 8; ++r) {
        int t = lane + 32 * r;
        ev[r] = ebuf[b * 256 + t];
        if (t < lenb && ev[r] > m) m = ev[r];
      }
#pragma unroll
      for (int off = 16; off >= 1; off >>= 1) m = fmaxf(m, __shfl_xor(m, off, 32));
      float sum = 0.f;
#pragma unroll
      for (int r = 0; r < 8; ++r) {
        int t = lane + 32 * r;
        float e = (t < lenb) ? __expf(ev[r] - m) : 0.f;
        ev[r] = e; sum += e;
      }
#pragma unroll
      for (int off = 16; off >= 1; off >>= 1) sum += __shfl_xor(sum, off, 32);
      float inv = 1.f / sum;
#pragma unroll
      for (int r = 0; r < 8; ++r) {
        int t = lane + 32 * r;
        float a = ev[r] * inv;
        ebuf[b * 256 + t] = a;
        attns[((long)s * 32 + b) * 256 + t] = a;
      }
    }
    __syncthreads();
    // context[b][v] = sum_t attn[b][t] * values[b][t][v]
#pragma unroll
    for (int r = 0; r < 4; ++r) {
      int idx = tid + r * 1024;
      int b = idx >> 7, v = idx & 127;
      const float* vp = values + (long)b * 256 * 128 + v;
      float a = 0.f;
      for (int t = 0; t < 256; ++t) a += ebuf[b * 256 + t] * vp[t * 128];
      bf16_t cb = f2bf(a);
      A1[b * 640 + 256 + v] = cb;
      Ap[b * 256 + 128 + v] = cb;
    }
    __syncthreads();
    // preds(32x1000) = Ap(32x256) @ WoutT^T + out_b   (126 tiles)
    for (int tt = wave; tt < 126; tt += 32) {
      int mtile = tt / 63, tn = tt % 63;
      int mrow = mtile * 16 + l15;
      v8f acc = vzero8();
      for (int k0 = 0; k0 < 256; k0 += 32) {
        v8bf a0 = *(const v8bf*)&Ap[mrow * 256 + k0 + half * 8];
        v8bf a1 = *(const v8bf*)&Ap[mrow * 256 + k0 + 16 + half * 8];
        v16bf av = cat8(a0, a1);
        v16bf bv = cat8(*(const v8bf*)(WoutT + (long)(tn * 16 + l15) * 256 + half * 8 + k0), *(const v8bf*)(WoutT + (long)(tn * 16 + l15) * 256 + 16 + half * 8 + k0));
        acc = __builtin_amdgcn_wmma_f32_16x16x32_bf16(false, av, false, bv, (short)0, acc, false, false);
      }
      int n = tn * 16 + l15;
      if (n < 1000) {
        float bb = outb[n];
#pragma unroll
        for (int r = 0; r < 8; ++r) {
          int b = mtile * 16 + half * 8 + r;
          preds[((long)b * 199 + s) * 1000 + n] = acc[r] + bb;
        }
      }
    }
    __syncthreads();
  }
}

// ---------------------------------------------------------------------------
extern "C" void kernel_launch(void* const* d_in, const int* in_sizes, int n_in,
                              void* d_out, int out_size, void* d_ws, size_t ws_size,
                              hipStream_t stream) {
  (void)in_sizes; (void)n_in; (void)out_size; (void)ws_size;
  const float* x     = (const float*)d_in[0];
  const int*   lens  = (const int*)d_in[1];
  const int*   text  = (const int*)d_in[2];
  const float* emb   = (const float*)d_in[3];
  const float* e0Wih = (const float*)d_in[4];
  const float* e0Whh = (const float*)d_in[5];
  const float* e0b   = (const float*)d_in[6];
  const float* pWih  = (const float*)d_in[7];
  const float* pWhh  = (const float*)d_in[8];
  const float* pb    = (const float*)d_in[9];
  const float* keyW  = (const float*)d_in[10];
  const float* keyb  = (const float*)d_in[11];
  const float* valW  = (const float*)d_in[12];
  const float* valb  = (const float*)d_in[13];
  const float* d1Wih = (const float*)d_in[14];
  const float* d1Whh = (const float*)d_in[15];
  const float* d1b   = (const float*)d_in[16];
  const float* d2Wih = (const float*)d_in[17];
  const float* d2Whh = (const float*)d_in[18];
  const float* d2b   = (const float*)d_in[19];
  const float* outW  = (const float*)d_in[20];
  const float* outb  = (const float*)d_in[21];
  float* preds = (float*)d_out;
  float* attns = preds + (long)32 * 199 * 1000;

  char* wsb = (char*)d_ws;
  size_t off = 0;
  auto carve = [&](size_t bytes) -> char* {
    char* p = wsb + off;
    off = (off + bytes + 255) & ~(size_t)255;
    return p;
  };
  bf16_t* xbf    = (bf16_t*)carve((size_t)32 * 2048 * 64 * 2);
  bf16_t* Wih0   = (bf16_t*)carve((size_t)2 * 1024 * 64 * 2);
  bf16_t* Whh0   = (bf16_t*)carve((size_t)2 * 1024 * 256 * 2);
  bf16_t* pWihb  = (bf16_t*)carve((size_t)6144 * 1024 * 2);
  bf16_t* pWhhb  = (bf16_t*)carve((size_t)6144 * 256 * 2);
  bf16_t* keyWb  = (bf16_t*)carve((size_t)128 * 512 * 2);
  bf16_t* valWb  = (bf16_t*)carve((size_t)128 * 512 * 2);
  bf16_t* Wc1    = (bf16_t*)carve((size_t)1024 * 640 * 2);
  bf16_t* Wc2    = (bf16_t*)carve((size_t)512 * 384 * 2);
  bf16_t* WoutT  = (bf16_t*)carve((size_t)1008 * 256 * 2);
  bf16_t* embbf  = (bf16_t*)carve((size_t)1000 * 256 * 2);
  float*  xWbuf  = (float*)carve((size_t)2 * 2048 * 32 * 1024 * 4);
  bf16_t* h0     = (bf16_t*)carve((size_t)32 * 2048 * 512 * 2);
  bf16_t* h1     = (bf16_t*)carve((size_t)32 * 1024 * 512 * 2);
  bf16_t* h2b    = (bf16_t*)carve((size_t)32 * 512 * 512 * 2);
  bf16_t* h3     = (bf16_t*)carve((size_t)32 * 256 * 512 * 2);
  float*  keys   = (float*)carve((size_t)32 * 256 * 128 * 4);
  float*  values = (float*)carve((size_t)32 * 256 * 128 * 4);

  auto gblk = [](long total) { return (unsigned)((total + 255) / 256); };

  // weight / input conversions
  cvt_x_k<<<gblk((long)32 * 2048 * 64), 256, 0, stream>>>(x, xbf, (long)32 * 2048);
  cvt_rows_k<<<gblk((long)2048 * 64), 256, 0, stream>>>(e0Wih, Wih0, 2048, 40, 64, 0, 64);
  cvt_rows_k<<<gblk((long)2048 * 256), 256, 0, stream>>>(e0Whh, Whh0, 2048, 256, 256, 0, 256);
  cvt_rows_k<<<gblk((long)6144 * 1024), 256, 0, stream>>>(pWih, pWihb, 6144, 1024, 1024, 0, 1024);
  cvt_rows_k<<<gblk((long)6144 * 256), 256, 0, stream>>>(pWhh, pWhhb, 6144, 256, 256, 0, 256);
  cvt_T_k<<<gblk((long)128 * 512), 256, 0, stream>>>(keyW, keyWb, 512, 128, 128);
  cvt_T_k<<<gblk((long)128 * 512), 256, 0, stream>>>(valW, valWb, 512, 128, 128);
  cvt_T_k<<<gblk((long)1008 * 256), 256, 0, stream>>>(outW, WoutT, 256, 1000, 1008);
  cvt_rows_k<<<gblk((long)1024 * 384), 256, 0, stream>>>(d1Wih, Wc1, 1024, 384, 384, 0, 640);
  cvt_rows_k<<<gblk((long)1024 * 256), 256, 0, stream>>>(d1Whh, Wc1, 1024, 256, 256, 384, 640);
  cvt_rows_k<<<gblk((long)512 * 256), 256, 0, stream>>>(d2Wih, Wc2, 512, 256, 256, 0, 384);
  cvt_rows_k<<<gblk((long)512 * 128), 256, 0, stream>>>(d2Whh, Wc2, 512, 128, 128, 256, 384);
  cvt_rows_k<<<gblk((long)1000 * 256), 256, 0, stream>>>(emb, embbf, 1000, 256, 256, 0, 256);

  // zero h buffers (invalid timesteps must read as 0)
  hipMemsetAsync(h0, 0, (size_t)32 * 2048 * 512 * 2, stream);
  hipMemsetAsync(h1, 0, (size_t)32 * 1024 * 512 * 2, stream);
  hipMemsetAsync(h2b, 0, (size_t)32 * 512 * 512 * 2, stream);
  hipMemsetAsync(h3, 0, (size_t)32 * 256 * 512 * 2, stream);

  auto gemm = [&](const bf16_t* A, const bf16_t* W, const float* bias, float* out,
                  int M, int N, int Kp, int Bdim, long sB, long sT) {
    int tilesN = N / 16;
    long total = (long)(M / 16) * tilesN;
    unsigned blocks = (unsigned)((total + 7) / 8);
    gemm_xw_k<<<blocks, 256, 0, stream>>>(A, W, bias, out, tilesN, (int)total, N, Kp, Bdim, sB, sT);
  };

  // ---- encoder layer 0 ----
  for (int d = 0; d < 2; ++d)
    gemm(xbf, Wih0 + (long)d * 1024 * 64, e0b + (long)d * 1024,
         xWbuf + (long)d * 2048 * 32 * 1024,
         65536, 1024, 64, 32, (long)2048 * 64, 64);
  lstm_rec_k<<<2, 1024, 0, stream>>>(xWbuf, Whh0, lens, 0, h0, 2048);

  // ---- pyramid layers 1..3 ----
  bf16_t* houts[4] = {h0, h1, h2b, h3};
  for (int l = 1; l <= 3; ++l) {
    int Tl = 2048 >> l, Tprev = 2048 >> (l - 1);
    bf16_t* hin = houts[l - 1];
    for (int d = 0; d < 2; ++d)
      gemm(hin, pWihb + (long)((l - 1) * 2 + d) * 1024 * 1024,
           pb + (long)((l - 1) * 2 + d) * 1024,
           xWbuf + (long)d * Tl * 32 * 1024,
           Tl * 32, 1024, 1024, 32, (long)Tprev * 512, 1024);
    lstm_rec_k<<<2, 1024, 0, stream>>>(xWbuf, pWhhb + (long)(l - 1) * 2 * 1024 * 256,
                                       lens, l, houts[l], Tl);
  }

  // ---- keys / values projection ----
  gemm(h3, keyWb, keyb, keys, 8192, 128, 512, 1, 0, 512);
  gemm(h3, valWb, valb, values, 8192, 128, 512, 1, 0, 512);

  // ---- attention decoder ----
  decoder_k<<<1, 1024, 0, stream>>>(text, embbf, keys, values, lens,
                                    Wc1, d1b, Wc2, d2b, WoutT, outb, preds, attns);
}
